// XTEembedding_90220083020259
// MI455X (gfx1250) — matrix operand order repaired and checked by reading, not compiled, not run on hardware
//
#include <hip/hip_runtime.h>
#include <stdint.h>

// CDNA5 WMMA operand types (wave32): 16 bf16 = 8 VGPRs (A/B), 8 f32 (C/D)
typedef __attribute__((ext_vector_type(16))) __bf16 v16bf;
typedef __attribute__((ext_vector_type(8)))  float  v8f;

#define WMMA_BF16(A, B, C) \
  __builtin_amdgcn_wmma_f32_16x16x32_bf16(false, (A), false, (B), (short)0, (C), false, false)

constexpr int B_ = 8, P_ = 144, L_ = 288, N_ = 512, D_ = 64;

// Module-scope scratch (no d_ws assumptions): recomputed every launch, read after
// same-stream ordering, so deterministic and graph-capture safe.
__device__ __align__(256) float  g_tb[B_ * L_ * D_];              // te @ W1[64:] + b1 (exact f32)
__device__ __align__(256) __bf16 g_wf[2 * 4 * 2 * 2 * 32 * 16];   // [mat][jt][kk][split][lane][16]

static __device__ __forceinline__ int wf_idx(int mat, int jt, int kk, int split, int lane) {
  return ((((mat * 4 + jt) * 2 + kk) * 2 + split) * 32 + lane) * 16;
}

// Native bf16 split: hi = rne(f), lo = rne(f - hi). gfx1250 true-16 cvt => ~4 VALU total.
static __device__ __forceinline__ void split_bf16(float f, __bf16& h, __bf16& l) {
  h = (__bf16)f;
  l = (__bf16)(f - (float)h);
}

// ---------- prologue 1: pre-swizzle W1[0:64] and W2 into per-lane WMMA B-fragments (hi/lo)
// B-frag layout (16-bit 32x16): lane(half,n): element i -> K = kk*32 + (i&7) + half*8 + (i>>3)*16, col = jt*16+n
__global__ __launch_bounds__(256) void prep_wf_kernel(const float* __restrict__ W1,
                                                      const float* __restrict__ W2) {
  const int t = blockIdx.x * 256 + threadIdx.x;
  if (t >= 512) return;                       // 2 mat * 4 jt * 2 kk * 32 lanes
  const int lane = t & 31;
  const int kk   = (t >> 5) & 1;
  const int jt   = (t >> 6) & 3;
  const int mat  = (t >> 8) & 1;
  const float* W = mat ? W2 : W1;             // W1a = first 64 rows of row-major [128,64]
  const int half = lane >> 4, hl = lane & 15;
  __bf16* hi = g_wf + wf_idx(mat, jt, kk, 0, lane);
  __bf16* lo = g_wf + wf_idx(mat, jt, kk, 1, lane);
  #pragma unroll
  for (int i = 0; i < 16; ++i) {
    const int K = kk * 32 + (i & 7) + half * 8 + ((i >> 3) << 4);
    const float wv = W[K * 64 + jt * 16 + hl];
    split_bf16(wv, hi[i], lo[i]);
  }
}

// ---------- prologue 2: exact f32 time-bias rows tb[b,l] = te(b,l) @ W1[64:128] + b1
__global__ __launch_bounds__(64) void prep_tb_kernel(const long long* __restrict__ TE,
                                                     const float* __restrict__ W_te,
                                                     const float* __restrict__ b_te,
                                                     const float* __restrict__ W1,
                                                     const float* __restrict__ b1) {
  const int bl = blockIdx.x;                  // B*L = 2304
  const int j  = threadIdx.x;                 // 64
  __shared__ float te_s[64];
  const long long t0 = TE[(size_t)bl * 2 + 0];
  const long long t1 = TE[(size_t)bl * 2 + 1];
  const int dow = (int)(t0 % 7);
  const int tod = (int)(t1 % 288);
  te_s[j] = W_te[dow * 64 + j] + W_te[(7 + tod) * 64 + j] + b_te[j];
  __syncthreads();
  float acc = b1[j];
  #pragma unroll 8
  for (int k = 0; k < 64; ++k)
    acc = fmaf(te_s[k], W1[(64 + k) * 64 + j], acc);
  g_tb[bl * 64 + j] = acc;
}

// ---------- main fused kernel
__global__ __launch_bounds__(256) void fused_te_mlp_kernel(const float* __restrict__ X,
                                                           const float* __restrict__ b2,
                                                           float* __restrict__ out) {
  __shared__ float Gs[64][68];   // padded stride -> conflict-free stripes
  __shared__ float tbs[2][64];

  const int blk = blockIdx.x;          // 8*144*8 = 9216 blocks
  const int nt  = blk & 7;             // 64-row slab of N
  const int p   = (blk >> 3) % P_;
  const int b   = blk / (8 * P_);

  const int tid  = threadIdx.x;
  const int lane = tid & 31;
  const int w    = tid >> 5;           // 8 waves
  const int half = lane >> 4;
  const int hl   = lane & 15;
  const int mt   = w & 3;              // 16-row stripe within the slab
  const int jtp  = w >> 2;             // column-tile pair {0,1} or {2,3}

  if (tid < 128) {
    const int v = tid >> 6, j = tid & 63;
    tbs[v][j] = g_tb[(size_t)((b * L_ + p + v * P_) * 64 + j)];
  }

  // ---- GEMM1: G(64x64) = X_tile @ W1[0:64]  (split-bf16, f32 accumulate)
  // A layout (16-bit 16x32): lane(half,m): element i -> K = (i&7) + half*8 + (i>>3)*16,
  // i.e. two contiguous 8-float runs at offsets {kk*32+half*8, kk*32+16+half*8}.
  const float* xrow = X + (((size_t)(b * P_ + p) * N_ + nt * 64 + mt * 16 + hl) * D_);
  v16bf ah[2], al[2];
  #pragma unroll
  for (int kk = 0; kk < 2; ++kk) {
    const int o0 = kk * 32 + half * 8;
    float f[16];
    const float4 r0 = *(const float4*)(xrow + o0);
    const float4 r1 = *(const float4*)(xrow + o0 + 4);
    const float4 r2 = *(const float4*)(xrow + o0 + 16);
    const float4 r3 = *(const float4*)(xrow + o0 + 20);
    f[0]=r0.x; f[1]=r0.y; f[2]=r0.z; f[3]=r0.w;
    f[4]=r1.x; f[5]=r1.y; f[6]=r1.z; f[7]=r1.w;
    f[8]=r2.x; f[9]=r2.y; f[10]=r2.z; f[11]=r2.w;
    f[12]=r3.x; f[13]=r3.y; f[14]=r3.z; f[15]=r3.w;
    #pragma unroll
    for (int i = 0; i < 16; ++i) {
      __bf16 h, l;
      split_bf16(f[i], h, l);
      ah[kk][i] = h;
      al[kk][i] = l;
    }
  }

  #pragma unroll
  for (int t = 0; t < 2; ++t) {
    const int jt = jtp * 2 + t;
    const v16bf bh0 = *(const v16bf*)(g_wf + wf_idx(0, jt, 0, 0, lane));
    const v16bf bl0 = *(const v16bf*)(g_wf + wf_idx(0, jt, 0, 1, lane));
    const v16bf bh1 = *(const v16bf*)(g_wf + wf_idx(0, jt, 1, 0, lane));
    const v16bf bl1 = *(const v16bf*)(g_wf + wf_idx(0, jt, 1, 1, lane));
    v8f c = {0.f, 0.f, 0.f, 0.f, 0.f, 0.f, 0.f, 0.f};
    c = WMMA_BF16(ah[0], bh0, c);
    c = WMMA_BF16(ah[0], bl0, c);
    c = WMMA_BF16(al[0], bh0, c);
    c = WMMA_BF16(ah[1], bh1, c);
    c = WMMA_BF16(ah[1], bl1, c);
    c = WMMA_BF16(al[1], bh1, c);
    // C layout: VGPR r -> row M = r + half*8, col = hl
    #pragma unroll
    for (int r = 0; r < 8; ++r)
      Gs[mt * 16 + half * 8 + r][jt * 16 + hl] = c[r];
  }

  __syncthreads();

  // ---- GEMM2 (both time copies): out = relu(G + tb[v]) @ W2 + b2
  #pragma unroll
  for (int v = 0; v < 2; ++v) {
    v16bf hh[2], hlo[2];
    #pragma unroll
    for (int kk = 0; kk < 2; ++kk) {
      #pragma unroll
      for (int i = 0; i < 16; ++i) {
        const int K = kk * 32 + (i & 7) + half * 8 + ((i >> 3) << 4);
        float g = Gs[mt * 16 + hl][K] + tbs[v][K];
        g = fmaxf(g, 0.0f);
        __bf16 h, l;
        split_bf16(g, h, l);
        hh[kk][i]  = h;
        hlo[kk][i] = l;
      }
    }
    const int l = p + v * P_;
    #pragma unroll
    for (int t = 0; t < 2; ++t) {
      const int jt = jtp * 2 + t;
      const v16bf bh0 = *(const v16bf*)(g_wf + wf_idx(1, jt, 0, 0, lane));
      const v16bf bl0 = *(const v16bf*)(g_wf + wf_idx(1, jt, 0, 1, lane));
      const v16bf bh1 = *(const v16bf*)(g_wf + wf_idx(1, jt, 1, 0, lane));
      const v16bf bl1 = *(const v16bf*)(g_wf + wf_idx(1, jt, 1, 1, lane));
      v8f c = {0.f, 0.f, 0.f, 0.f, 0.f, 0.f, 0.f, 0.f};
      c = WMMA_BF16(hh[0], bh0, c);
      c = WMMA_BF16(hh[0], bl0, c);
      c = WMMA_BF16(hlo[0], bh0, c);
      c = WMMA_BF16(hh[1], bh1, c);
      c = WMMA_BF16(hh[1], bl1, c);
      c = WMMA_BF16(hlo[1], bh1, c);
      const float bias = b2[jt * 16 + hl];
      const size_t obase =
          (((size_t)(b * L_ + l) * N_) + nt * 64 + mt * 16 + half * 8) * 64 + jt * 16 + hl;
      #pragma unroll
      for (int r = 0; r < 8; ++r)
        out[obase + (size_t)r * 64] = c[r] + bias;
    }
  }
}

extern "C" void kernel_launch(void* const* d_in, const int* in_sizes, int n_in,
                              void* d_out, int out_size, void* d_ws, size_t ws_size,
                              hipStream_t stream) {
  (void)in_sizes; (void)n_in; (void)out_size; (void)d_ws; (void)ws_size;
  const float*     X    = (const float*)d_in[0];
  const long long* TE   = (const long long*)d_in[1];
  const float*     W_te = (const float*)d_in[2];
  const float*     b_te = (const float*)d_in[3];
  const float*     W1   = (const float*)d_in[4];
  const float*     b1   = (const float*)d_in[5];
  const float*     W2   = (const float*)d_in[6];
  const float*     b2   = (const float*)d_in[7];
  float* out = (float*)d_out;

  // tiny prologues (same stream -> ordered before the fused kernel)
  prep_wf_kernel<<<2, 256, 0, stream>>>(W1, W2);
  prep_tb_kernel<<<B_ * L_, 64, 0, stream>>>(TE, W_te, b_te, W1, b1);

  // 8 batches * 144 time slots * (512/64) n-slabs
  fused_te_mlp_kernel<<<8 * 144 * 8, 256, 0, stream>>>(X, b2, out);
}
